// ShortcutMoEDecoderLayer_88235808129203
// MI455X (gfx1250) — matrix operand order repaired and checked by reading, not compiled, not run on hardware
//
#include <hip/hip_runtime.h>
#include <hip/hip_bf16.h>

#define T_ 1024
#define H_ 1024
#define NQ_ 16
#define NK_ 4
#define D_ 64
#define QKV_ ((NQ_ + 2 * NK_) * D_) /* 1536 */
#define IDENSE_ 4096
#define IMOE_ 512
#define E_ 8

typedef __attribute__((ext_vector_type(16))) __bf16 v16bf;
typedef __attribute__((ext_vector_type(8))) __bf16 v8bf;
typedef __attribute__((ext_vector_type(8))) float v8f;
typedef __attribute__((ext_vector_type(4))) unsigned int v4u;
typedef __attribute__((ext_vector_type(8))) int v8i;
typedef __attribute__((ext_vector_type(4))) int v4i;

#if defined(__HIP_DEVICE_COMPILE__) &&                        \
    __has_builtin(__builtin_amdgcn_tensor_load_to_lds) &&     \
    __has_builtin(__builtin_amdgcn_s_wait_tensorcnt)
#define USE_TDM 1
#else
#define USE_TDM 0
#endif

static __device__ inline v8f wmma_bf16(v16bf a, v16bf b, v8f c) {
  return __builtin_amdgcn_wmma_f32_16x16x32_bf16(false, a, false, b, (short)0, c,
                                                 false, false);
}

#if USE_TDM
// 2D bf16 tile DMA: global (rowStride elems) -> LDS, tile_d0 x tile_d1, with
// LDS padding of 8 dwords every (2<<intervalCode) dwords.
static __device__ inline void tdm_load_tile_bf16(const __bf16* gsrc,
                                                 void* lds_dst, int rowStride,
                                                 int tile_d0, int tile_d1,
                                                 int intervalCode) {
  const unsigned long long ga = (unsigned long long)gsrc;
  const unsigned int lds = (unsigned int)(unsigned long long)lds_dst;
  v4u g0;
  g0[0] = 1u;  // count=1
  g0[1] = lds;
  g0[2] = (unsigned int)ga;
  g0[3] = (unsigned int)((ga >> 32) & 0x1FFFFFFull) | (2u << 30);  // type=2
  v8i g1;
  g1[0] = (1 << 16) | (1 << 20) | (intervalCode << 22) | (7 << 25);
  g1[1] = ((unsigned int)rowStride & 0xFFFFu) << 16;
  g1[2] = (int)((((unsigned int)rowStride >> 16) & 0xFFFFu) |
                (((unsigned int)T_ & 0xFFFFu) << 16));
  g1[3] = (int)(((unsigned int)tile_d0 & 0xFFFFu) << 16);
  g1[4] = (int)((unsigned int)tile_d1 & 0xFFFFu);
  g1[5] = rowStride;
  g1[6] = 0;
  g1[7] = 0;
  v4i z4 = {0, 0, 0, 0};
#if __clang_major__ >= 23
  v8i z8 = {0, 0, 0, 0, 0, 0, 0, 0};
  __builtin_amdgcn_tensor_load_to_lds(g0, g1, z4, z4, z8, 0);
#else
  __builtin_amdgcn_tensor_load_to_lds(g0, g1, z4, z4, 0);
#endif
}
#endif

// ---------------------------------------------------------------------------
__global__ void f32_to_bf16_kernel(const float* __restrict__ in,
                                   __bf16* __restrict__ out, size_t n) {
  for (size_t i = (size_t)blockIdx.x * blockDim.x + threadIdx.x; i < n;
       i += (size_t)gridDim.x * blockDim.x)
    out[i] = (__bf16)in[i];
}

// ---------------------------------------------------------------------------
// Fused residual add + RMSNorm.  h_out(bf16) = rmsnorm(x + res) * w,
// r_out(f32) = x + res.   res_in may alias r_out.
// ---------------------------------------------------------------------------
__global__ __launch_bounds__(256) void add_rmsnorm_kernel(
    const float* __restrict__ x, const float* __restrict__ res_in,
    const float* __restrict__ w, __bf16* __restrict__ h_out,
    float* __restrict__ r_out) {
  const int row = blockIdx.x;
  __shared__ float red[256];
  float acc = 0.f;
  for (int c = threadIdx.x; c < H_; c += 256) {
    float s = x[(size_t)row * H_ + c] + res_in[(size_t)row * H_ + c];
    r_out[(size_t)row * H_ + c] = s;
    acc += s * s;
  }
  red[threadIdx.x] = acc;
  __syncthreads();
  for (int off = 128; off; off >>= 1) {
    if (threadIdx.x < off) red[threadIdx.x] += red[threadIdx.x + off];
    __syncthreads();
  }
  const float rs = rsqrtf(red[0] / (float)H_ + 1e-6f);
  for (int c = threadIdx.x; c < H_; c += 256)
    h_out[(size_t)row * H_ + c] = (__bf16)(r_out[(size_t)row * H_ + c] * rs * w[c]);
}

// ---------------------------------------------------------------------------
// GEMM: C[M,N](f32) = A[M,K](bf16) * B[K,N](bf16) via WMMA, pipelined:
//  - A tile k+1 DMA'd by TDM into double-buffered LDS while tile k computes.
//  - B tile k+1 global-loaded into registers before compute of tile k and
//    ds_stored (transposed) at the next iteration top.
// 64x128 tile / 128-thread block; wave = 16 rows x 128 cols = 8 WMMAs/K-step.
// ---------------------------------------------------------------------------
__global__ __launch_bounds__(128) void gemm_bf16_wmma_kernel(
    const __bf16* __restrict__ A, const __bf16* __restrict__ B,
    float* __restrict__ C, int M, int N, int K,
    const float* __restrict__ rowScale, int scaleStride, int scaleOff,
    int accumulate) {
  __shared__ __bf16 As[2][64][48];  // [buf][m][k], stride 96B (TDM pad)
  __shared__ __bf16 BsT[128][48];   // [n][k], stride 96B
  const int tid = threadIdx.x;
  const int lane = tid & 31;
  const int w = tid >> 5;
  const int m0 = blockIdx.y * 64, n0 = blockIdx.x * 128;

  v8f acc[8];
#pragma unroll
  for (int c = 0; c < 8; c++)
#pragma unroll
    for (int i = 0; i < 8; i++) acc[c][i] = 0.f;

  const int br = tid >> 2, bc = (tid & 3) * 32;  // B staging coords
  const int am = w * 16 + (lane & 15);
  const int ko = (lane < 16) ? 0 : 8;
  const int kb = (lane < 16) ? 0 : 16;

  const __bf16* Bbase = B + (size_t)br * N + n0 + bc;
  auto loadB = [&](int k0, v8bf* r) {
    const v8bf* p = (const v8bf*)(Bbase + (size_t)k0 * N);
#pragma unroll
    for (int j = 0; j < 4; j++) r[j] = p[j];
  };
  auto storeB = [&](const v8bf* r) {
#pragma unroll
    for (int j = 0; j < 32; j++) BsT[bc + j][br] = r[j >> 3][j & 7];
  };
#if !USE_TDM
  auto stageA = [&](int k0, int b_) {
    const int ar = tid >> 1, ac = (tid & 1) * 16;
    const __bf16* Ap = A + (size_t)(m0 + ar) * K + k0 + ac;
#pragma unroll
    for (int j = 0; j < 16; j++) As[b_][ar][ac + j] = Ap[j];
  };
#endif

  v8bf bcur[4], bnxt[4];
  loadB(0, bcur);
#if USE_TDM
  if (tid < 32)
    tdm_load_tile_bf16(A + (size_t)m0 * K, &As[0][0][0], K, 32, 64, 3);
#else
  stageA(0, 0);
#endif
  int buf = 0;
  for (int k0 = 0; k0 < K; k0 += 32) {
    storeB(bcur);
#if USE_TDM
    if (tid < 32) __builtin_amdgcn_s_wait_tensorcnt(0);
#endif
    __syncthreads();
    const bool more = (k0 + 32) < K;
    if (more) {
      loadB(k0 + 32, bnxt);
#if USE_TDM
      if (tid < 32)
        tdm_load_tile_bf16(A + (size_t)m0 * K + k0 + 32, &As[buf ^ 1][0][0], K,
                           32, 64, 3);
#else
      stageA(k0 + 32, buf ^ 1);
#endif
      if (k0 + 64 < K)  // prime L2 two tiles ahead -> global_prefetch_b8
        __builtin_prefetch((const void*)(Bbase + (size_t)(k0 + 64) * N), 0, 1);
    }

    v16bf af;  // A fragment 16x32
#pragma unroll
    for (int j = 0; j < 8; j++) {
      af[j] = As[buf][am][kb + j];
      af[8 + j] = As[buf][am][kb + 8 + j];
    }
#pragma unroll
    for (int c = 0; c < 8; c++) {
      v16bf bfv;  // B fragment: contiguous 16 elems from BsT row
#pragma unroll
      for (int e = 0; e < 16; e++) bfv[e] = BsT[c * 16 + (lane & 15)][kb + e];
      acc[c] = wmma_bf16(af, bfv, acc[c]);
    }
    __syncthreads();
#pragma unroll
    for (int j = 0; j < 4; j++) bcur[j] = bnxt[j];
    buf ^= 1;
  }

#pragma unroll
  for (int c = 0; c < 8; c++) {
#pragma unroll
    for (int i = 0; i < 8; i++) {
      const int m = m0 + w * 16 + i + ((lane >> 4) << 3);
      const int n = n0 + c * 16 + (lane & 15);
      const size_t idx = (size_t)m * N + n;
      float v = acc[c][i];
      if (rowScale) v *= rowScale[(size_t)m * scaleStride + scaleOff];
      if (accumulate)
        C[idx] += v;
      else
        C[idx] = v;
    }
  }
}

// ---------------------------------------------------------------------------
// RoPE + f32->bf16 for the whole QKV buffer (rope on q/k, passthrough on v).
// ---------------------------------------------------------------------------
__global__ __launch_bounds__(256) void rope_bf16_kernel(
    const float* __restrict__ qkv, __bf16* __restrict__ qkv_b,
    const int* __restrict__ pos) {
  const int t = blockIdx.x;
  const float p = (float)pos[t];
  const float* src = qkv + (size_t)t * QKV_;
  __bf16* dst = qkv_b + (size_t)t * QKV_;
  for (int idx = threadIdx.x; idx < (NQ_ + NK_) * 32; idx += 256) {
    const int head = idx >> 5, j = idx & 31;
    const int off = (head < NQ_) ? head * D_ : NQ_ * D_ + (head - NQ_) * D_;
    const float inv = __powf(10000.0f, -(float)j / 32.0f);
    const float f = p * inv;
    const float cs = __cosf(f), sn = __sinf(f);
    const float x1 = src[off + j], x2 = src[off + j + 32];
    dst[off + j] = (__bf16)(x1 * cs - x2 * sn);
    dst[off + j + 32] = (__bf16)(x2 * cs + x1 * sn);
  }
  for (int vi = threadIdx.x; vi < NK_ * D_; vi += 256)
    dst[(NQ_ + NK_) * D_ + vi] = (__bf16)src[(NQ_ + NK_) * D_ + vi];
}

// ---------------------------------------------------------------------------
// Flash attention on bf16 QKV, pipelined like the GEMM:
//  - K tile kt+1 via TDM into double-buffered LDS during compute of kt.
//  - V tile kt+1 into registers during compute, transposed-stored next top.
// 128-thread block = 4 waves = 4 query tiles of one head.
// ---------------------------------------------------------------------------
__global__ __launch_bounds__(128) void flash_attn_kernel(
    const __bf16* __restrict__ qkv, __bf16* __restrict__ out) {
  const int qtg = blockIdx.x;
  const int h = blockIdx.y;
  const int kvh = h >> 2;
  const int tid = threadIdx.x, lane = tid & 31, w = tid >> 5;
  const int qt = qtg * 4 + w;
  __shared__ __bf16 Ks[2][32][80];  // [buf][key][d], stride 160B (TDM pad)
  __shared__ __bf16 VsT[64][40];    // [d][key], stride 80B
  __shared__ __bf16 Ps[4][16][40];

  const int am = lane & 15;
  const int ko = (lane < 16) ? 0 : 8;
  const int kb = (lane < 16) ? 0 : 16;

  v16bf qa0, qa1;
  {
    const __bf16* Qp = qkv + (size_t)(qt * 16 + am) * QKV_ + h * D_;
#pragma unroll
    for (int j = 0; j < 8; j++) {
      qa0[j] = Qp[kb + j];
      qa0[8 + j] = Qp[kb + 8 + j];
      qa1[j] = Qp[32 + kb + j];
      qa1[8 + j] = Qp[32 + kb + 8 + j];
    }
  }

  v8f o[4];
#pragma unroll
  for (int c = 0; c < 4; c++)
#pragma unroll
    for (int i = 0; i < 8; i++) o[c][i] = 0.f;
  float mrun[8], lrun[8];
#pragma unroll
  for (int i = 0; i < 8; i++) {
    mrun[i] = -1e30f;
    lrun[i] = 0.f;
  }

  const int skk = tid >> 2, sc0 = (tid & 3) * 16;
  const __bf16* Vbase =
      qkv + (size_t)skk * QKV_ + (NQ_ + NK_) * D_ + kvh * D_ + sc0;
  auto loadV = [&](int kt, v8bf* r) {
    const v8bf* p = (const v8bf*)(Vbase + (size_t)(kt * 32) * QKV_);
    r[0] = p[0];
    r[1] = p[1];
  };
  auto storeV = [&](const v8bf* r) {
#pragma unroll
    for (int j = 0; j < 16; j++) VsT[sc0 + j][skk] = r[j >> 3][j & 7];
  };
#if !USE_TDM
  auto stageK = [&](int kt, int b_) {
    const __bf16* Kp =
        qkv + (size_t)(kt * 32 + skk) * QKV_ + NQ_ * D_ + kvh * D_ + sc0;
#pragma unroll
    for (int j = 0; j < 16; j++) Ks[b_][skk][sc0 + j] = Kp[j];
  };
#endif

  const int ktmax = ((qtg * 4 + 3) * 16 + 15) >> 5;
  v8bf vcur[2], vnxt[2];
  loadV(0, vcur);
#if USE_TDM
  if (tid < 32)
    tdm_load_tile_bf16(qkv + NQ_ * D_ + kvh * D_, &Ks[0][0][0], QKV_, 64, 32, 4);
#else
  stageK(0, 0);
#endif
  int buf = 0;
  for (int kt = 0; kt <= ktmax; kt++) {
    storeV(vcur);
#if USE_TDM
    if (tid < 32) __builtin_amdgcn_s_wait_tensorcnt(0);
#endif
    __syncthreads();
    const bool more = kt < ktmax;
    if (more) {
      loadV(kt + 1, vnxt);
#if USE_TDM
      if (tid < 32)
        tdm_load_tile_bf16(
            qkv + (size_t)((kt + 1) * 32) * QKV_ + NQ_ * D_ + kvh * D_,
            &Ks[buf ^ 1][0][0], QKV_, 64, 32, 4);
#else
      stageK(kt + 1, buf ^ 1);
#endif
    }

    v8f s0, s1;
#pragma unroll
    for (int i = 0; i < 8; i++) {
      s0[i] = 0.f;
      s1[i] = 0.f;
    }
    {
      v16bf b0, b1, b2, b3;
#pragma unroll
      for (int e = 0; e < 16; e++) {
        b0[e] = Ks[buf][am][kb + e];
        b1[e] = Ks[buf][am][32 + kb + e];
        b2[e] = Ks[buf][16 + am][kb + e];
        b3[e] = Ks[buf][16 + am][32 + kb + e];
      }
      s0 = wmma_bf16(qa0, b0, s0);
      s0 = wmma_bf16(qa1, b1, s0);
      s1 = wmma_bf16(qa0, b2, s1);
      s1 = wmma_bf16(qa1, b3, s1);
    }

#pragma unroll
    for (int i = 0; i < 8; i++) {
      const int m = i + ((lane >> 4) << 3);
      const int qrow = qt * 16 + m;
      const int key0 = kt * 32 + (lane & 15);
      const int key1 = key0 + 16;
      float v0 = s0[i] * 0.125f;
      float v1 = s1[i] * 0.125f;
      if (key0 > qrow) v0 = -1e30f;
      if (key1 > qrow) v1 = -1e30f;
      float rmax = fmaxf(v0, v1);
      for (int off = 8; off; off >>= 1)
        rmax = fmaxf(rmax, __shfl_xor(rmax, off, 16));
      const float mnew = fmaxf(mrun[i], rmax);
      const float corr = __expf(mrun[i] - mnew);
      const float p0 = __expf(v0 - mnew);
      const float p1 = __expf(v1 - mnew);
      float psum = p0 + p1;
      for (int off = 8; off; off >>= 1) psum += __shfl_xor(psum, off, 16);
      lrun[i] = lrun[i] * corr + psum;
      mrun[i] = mnew;
#pragma unroll
      for (int c = 0; c < 4; c++) o[c][i] *= corr;
      Ps[w][m][lane & 15] = (__bf16)p0;
      Ps[w][m][16 + (lane & 15)] = (__bf16)p1;
    }
    __syncthreads();

    v16bf pf;
#pragma unroll
    for (int j = 0; j < 8; j++) {
      pf[j] = Ps[w][am][kb + j];
      pf[8 + j] = Ps[w][am][kb + 8 + j];
    }
#pragma unroll
    for (int c = 0; c < 4; c++) {
      v16bf vb;
#pragma unroll
      for (int e = 0; e < 16; e++) vb[e] = VsT[c * 16 + (lane & 15)][kb + e];
      o[c] = wmma_bf16(pf, vb, o[c]);
    }
    __syncthreads();
    vcur[0] = vnxt[0];
    vcur[1] = vnxt[1];
    buf ^= 1;
  }

#pragma unroll
  for (int i = 0; i < 8; i++) {
    const int m = i + ((lane >> 4) << 3);
    const float inv = 1.0f / lrun[i];
#pragma unroll
    for (int c = 0; c < 4; c++)
      out[(size_t)(qt * 16 + m) * (NQ_ * D_) + h * D_ + c * 16 + (lane & 15)] =
          (__bf16)(o[c][i] * inv);
  }
}

// ---------------------------------------------------------------------------
// MoE gate on bf16 activations: one wave per token -> softmax -> top-2.
// ---------------------------------------------------------------------------
__global__ __launch_bounds__(256) void moe_gate_kernel(
    const __bf16* __restrict__ x, const float* __restrict__ gw,
    float* __restrict__ combine) {
  const int wave = threadIdx.x >> 5, lane = threadIdx.x & 31;
  const int row = blockIdx.x * 8 + wave;
  float acc[E_];
#pragma unroll
  for (int e = 0; e < E_; e++) acc[e] = 0.f;
  for (int c = lane; c < H_; c += 32) {
    const float xv = (float)x[(size_t)row * H_ + c];
#pragma unroll
    for (int e = 0; e < E_; e++) acc[e] += xv * gw[(size_t)c * E_ + e];
  }
#pragma unroll
  for (int e = 0; e < E_; e++)
    for (int off = 16; off; off >>= 1) acc[e] += __shfl_xor(acc[e], off, 32);
  if (lane == 0) {
    float mx = acc[0];
    for (int e = 1; e < E_; e++) mx = fmaxf(mx, acc[e]);
    float p[E_], s = 0.f;
    for (int e = 0; e < E_; e++) {
      p[e] = __expf(acc[e] - mx);
      s += p[e];
    }
    for (int e = 0; e < E_; e++) p[e] /= s;
    int i1 = 0;
    for (int e = 1; e < E_; e++)
      if (p[e] > p[i1]) i1 = e;
    const float w1 = p[i1];
    p[i1] = -1.f;
    int i2 = 0;
    for (int e = 1; e < E_; e++)
      if (p[e] > p[i2]) i2 = e;
    const float w2 = p[i2];
    for (int e = 0; e < E_; e++)
      combine[(size_t)row * E_ + e] = (e == i1) ? w1 : ((e == i2) ? w2 : 0.f);
  }
}

// ---------------------------------------------------------------------------
__global__ void silu_mul_kernel(const float* __restrict__ g,
                                __bf16* __restrict__ out, int rows, int I) {
  const size_t total = (size_t)rows * I;
  for (size_t idx = (size_t)blockIdx.x * blockDim.x + threadIdx.x; idx < total;
       idx += (size_t)gridDim.x * blockDim.x) {
    const size_t r = idx / I, c = idx % I;
    const float gate = g[r * (size_t)(2 * I) + c];
    const float up = g[r * (size_t)(2 * I) + I + c];
    out[idx] = (__bf16)((gate / (1.f + __expf(-gate))) * up);
  }
}

__global__ void fill_zero_kernel(float* __restrict__ p, size_t n) {
  for (size_t i = (size_t)blockIdx.x * blockDim.x + threadIdx.x; i < n;
       i += (size_t)gridDim.x * blockDim.x)
    p[i] = 0.f;
}

__global__ void add_kernel(const float* __restrict__ a,
                           const float* __restrict__ b, float* __restrict__ c,
                           size_t n) {
  for (size_t i = (size_t)blockIdx.x * blockDim.x + threadIdx.x; i < n;
       i += (size_t)gridDim.x * blockDim.x)
    c[i] = a[i] + b[i];
}

// ---------------------------------------------------------------------------
extern "C" void kernel_launch(void* const* d_in, const int* in_sizes, int n_in,
                              void* d_out, int out_size, void* d_ws,
                              size_t ws_size, hipStream_t stream) {
  (void)in_sizes;
  (void)n_in;
  (void)out_size;
  (void)ws_size;
  const float* hidden = (const float*)d_in[0];
  const float* residual = (const float*)d_in[1];
  const int* positions = (const int*)d_in[2];
  const float* ln0_w = (const float*)d_in[3];
  const float* pa0_w = (const float*)d_in[4];
  const float* ln1_w = (const float*)d_in[5];
  const float* pa1_w = (const float*)d_in[6];
  const float* gate_w = (const float*)d_in[15];

  float* out = (float*)d_out;
  float* r = out + (size_t)T_ * H_;

  char* cur = (char*)d_ws;
  auto alloc = [&](size_t bytes) {
    void* p = cur;
    cur += (bytes + 255) & ~(size_t)255;
    return p;
  };
  __bf16* h_b = (__bf16*)alloc((size_t)T_ * H_ * 2);
  float* qkv_f = (float*)alloc((size_t)T_ * QKV_ * 4);
  __bf16* qkv_b = (__bf16*)alloc((size_t)T_ * QKV_ * 2);
  __bf16* attn_b = (__bf16*)alloc((size_t)T_ * NQ_ * D_ * 2);
  float* tmp = (float*)alloc((size_t)T_ * H_ * 4);
  float* gu = (float*)alloc((size_t)T_ * 2 * IDENSE_ * 4);
  __bf16* act_b = (__bf16*)alloc((size_t)T_ * IDENSE_ * 2);
  float* shortcut = (float*)alloc((size_t)T_ * H_ * 4);
  float* combine = (float*)alloc((size_t)T_ * E_ * 4);
  float* moe_g = (float*)alloc((size_t)T_ * 2 * IMOE_ * 4);
  __bf16* moe_hb = (__bf16*)alloc((size_t)T_ * IMOE_ * 2);
  __bf16* qkv0_wb = (__bf16*)alloc((size_t)H_ * QKV_ * 2);
  __bf16* o0_wb = (__bf16*)alloc((size_t)NQ_ * D_ * H_ * 2);
  __bf16* qkv1_wb = (__bf16*)alloc((size_t)H_ * QKV_ * 2);
  __bf16* o1_wb = (__bf16*)alloc((size_t)NQ_ * D_ * H_ * 2);
  __bf16* gu0_wb = (__bf16*)alloc((size_t)H_ * 2 * IDENSE_ * 2);
  __bf16* dn0_wb = (__bf16*)alloc((size_t)IDENSE_ * H_ * 2);
  __bf16* gu1_wb = (__bf16*)alloc((size_t)H_ * 2 * IDENSE_ * 2);
  __bf16* dn1_wb = (__bf16*)alloc((size_t)IDENSE_ * H_ * 2);
  __bf16* w13_b = (__bf16*)alloc((size_t)E_ * H_ * 2 * IMOE_ * 2);
  __bf16* w2_b = (__bf16*)alloc((size_t)E_ * IMOE_ * H_ * 2);

  auto cvt = [&](const void* src, __bf16* dst, size_t n) {
    hipLaunchKernelGGL(f32_to_bf16_kernel, dim3(2048), dim3(256), 0, stream,
                       (const float*)src, dst, n);
  };
  cvt(d_in[7], qkv0_wb, (size_t)H_ * QKV_);
  cvt(d_in[8], o0_wb, (size_t)NQ_ * D_ * H_);
  cvt(d_in[9], qkv1_wb, (size_t)H_ * QKV_);
  cvt(d_in[10], o1_wb, (size_t)NQ_ * D_ * H_);
  cvt(d_in[11], gu0_wb, (size_t)H_ * 2 * IDENSE_);
  cvt(d_in[12], dn0_wb, (size_t)IDENSE_ * H_);
  cvt(d_in[13], gu1_wb, (size_t)H_ * 2 * IDENSE_);
  cvt(d_in[14], dn1_wb, (size_t)IDENSE_ * H_);
  cvt(d_in[16], w13_b, (size_t)E_ * H_ * 2 * IMOE_);
  cvt(d_in[17], w2_b, (size_t)E_ * IMOE_ * H_);

  auto gemm = [&](const __bf16* A, const __bf16* B, float* C, int M, int N,
                  int K, const float* scale, int sstride, int soff, int accum) {
    dim3 grid(N / 128, M / 64);
    hipLaunchKernelGGL(gemm_bf16_wmma_kernel, grid, dim3(128), 0, stream, A, B,
                       C, M, N, K, scale, sstride, soff, accum);
  };
  auto attention = [&](const __bf16* qkv_w, const __bf16* o_w) {
    gemm(h_b, qkv_w, qkv_f, T_, QKV_, H_, nullptr, 0, 0, 0);
    hipLaunchKernelGGL(rope_bf16_kernel, dim3(T_), dim3(256), 0, stream, qkv_f,
                       qkv_b, positions);
    hipLaunchKernelGGL(flash_attn_kernel, dim3(T_ / 64, NQ_), dim3(128), 0,
                       stream, qkv_b, attn_b);
    gemm(attn_b, o_w, tmp, T_, H_, NQ_ * D_, nullptr, 0, 0, 0);
  };

  hipLaunchKernelGGL(add_rmsnorm_kernel, dim3(T_), dim3(256), 0, stream, hidden,
                     residual, ln0_w, h_b, r);
  attention(qkv0_wb, o0_wb);
  hipLaunchKernelGGL(add_rmsnorm_kernel, dim3(T_), dim3(256), 0, stream, tmp, r,
                     pa0_w, h_b, r);
  hipLaunchKernelGGL(moe_gate_kernel, dim3(T_ / 8), dim3(256), 0, stream, h_b,
                     gate_w, combine);
  hipLaunchKernelGGL(fill_zero_kernel, dim3(1024), dim3(256), 0, stream,
                     shortcut, (size_t)T_ * H_);
  for (int e = 0; e < E_; e++) {
    gemm(h_b, w13_b + (size_t)e * H_ * 2 * IMOE_, moe_g, T_, 2 * IMOE_, H_,
         nullptr, 0, 0, 0);
    hipLaunchKernelGGL(silu_mul_kernel, dim3(1024), dim3(256), 0, stream, moe_g,
                       moe_hb, T_, IMOE_);
    gemm(moe_hb, w2_b + (size_t)e * IMOE_ * H_, shortcut, T_, H_, IMOE_,
         combine, E_, e, 1);
  }
  gemm(h_b, gu0_wb, gu, T_, 2 * IDENSE_, H_, nullptr, 0, 0, 0);
  hipLaunchKernelGGL(silu_mul_kernel, dim3(2048), dim3(256), 0, stream, gu,
                     act_b, T_, IDENSE_);
  gemm(act_b, dn0_wb, tmp, T_, H_, IDENSE_, nullptr, 0, 0, 0);
  hipLaunchKernelGGL(add_rmsnorm_kernel, dim3(T_), dim3(256), 0, stream, tmp, r,
                     ln1_w, h_b, r);
  attention(qkv1_wb, o1_wb);
  hipLaunchKernelGGL(add_rmsnorm_kernel, dim3(T_), dim3(256), 0, stream, tmp, r,
                     pa1_w, h_b, r);
  gemm(h_b, gu1_wb, gu, T_, 2 * IDENSE_, H_, nullptr, 0, 0, 0);
  hipLaunchKernelGGL(silu_mul_kernel, dim3(2048), dim3(256), 0, stream, gu,
                     act_b, T_, IDENSE_);
  gemm(act_b, dn1_wb, tmp, T_, H_, IDENSE_, nullptr, 0, 0, 0);
  hipLaunchKernelGGL(add_kernel, dim3(1024), dim3(256), 0, stream, tmp,
                     shortcut, out, (size_t)T_ * H_);
}
